// Attention_11218454577761
// MI455X (gfx1250) — compile-verified, matrix operand order checked by reading, not run on hardware
//
#include <hip/hip_runtime.h>
#include <hip/hip_bf16.h>
#include <math.h>

// Problem constants (match reference)
#define B_    4
#define S_    2048
#define D_    2048
#define H_    16
#define KVH_  8
#define HD_   128
#define NREP_ 2

typedef __bf16 bf16;
typedef unsigned int u32;
typedef unsigned long long u64;
typedef __attribute__((ext_vector_type(16))) __bf16 v16bf;
typedef __attribute__((ext_vector_type(8)))  __bf16 v8bf;
typedef __attribute__((ext_vector_type(8)))  float  v8f;
typedef __attribute__((ext_vector_type(4)))  unsigned int v4u;
typedef __attribute__((ext_vector_type(8)))  int v8i;
typedef __attribute__((ext_vector_type(4)))  int v4i;

// ---------------------------------------------------------------- TDM helper
// Issue a Tensor-Data-Mover 2D tile load: rows x cols bf16 elements from a
// row-major global matrix (row stride gstride elements) into LDS, packed as
// [rows][cols]. Descriptor per CDNA5 ISA (D# groups 0/1).
// Tracked by TENSORcnt; caller must s_wait_tensorcnt before consuming.
#if __has_include(<hip/amd_detail/amd_gfx1250_TDM.h>)
#define TDM_LOAD_CALL(g0, g1, z4, z8) \
  __builtin_amdgcn_tensor_load_to_lds(g0, g1, z4, z4, z8, 0)
#else
#define TDM_LOAD_CALL(g0, g1, z4, z8) \
  __builtin_amdgcn_tensor_load_to_lds(g0, g1, z4, z4, 0)
#endif

static __device__ inline void tdm_load_2d(const bf16* gsrc, void* lds_dst,
                                          u32 rows, u32 cols, u32 gstride) {
  u32 lds = (u32)(uintptr_t)lds_dst;
  u64 ga  = (u64)(uintptr_t)gsrc;
  v4u g0;
  g0[0] = 1u;                                        // count=1 (valid user D#)
  g0[1] = lds;                                       // lds_addr
  g0[2] = (u32)(ga & 0xFFFFFFFFull);                 // global_addr[31:0]
  g0[3] = (u32)((ga >> 32) & 0x1FFFFFFull)           // global_addr[56:32]
          | (2u << 30);                              // type=2 ("image")
  v8i g1;
  g1[0] = (int)(1u << 16);                           // data_size=1 -> 2 bytes
  g1[1] = (int)((cols & 0xFFFFu) << 16);             // tensor_dim0[15:0]
  g1[2] = (int)(((cols >> 16) & 0xFFFFu)             // tensor_dim0[31:16]
          | ((rows & 0xFFFFu) << 16));               // tensor_dim1[15:0]
  g1[3] = (int)(((rows >> 16) & 0xFFFFu)             // tensor_dim1[31:16]
          | ((cols & 0xFFFFu) << 16));               // tile_dim0 = cols
  g1[4] = (int)(rows & 0xFFFFu);                     // tile_dim1 = rows, tile_dim2=0
  g1[5] = (int)gstride;                              // tensor_dim0_stride[31:0]
  g1[6] = 0;                                         // stride hi / dim1_stride lo
  g1[7] = 0;
  v4i z4 = {0, 0, 0, 0};
  v8i z8 = {0, 0, 0, 0, 0, 0, 0, 0};
  (void)z8;
  TDM_LOAD_CALL(g0, g1, z4, z8);
}

// Load a 16-half WMMA fragment slice from an LDS row per ISA 7.12.2:
// elements 0..7  = row[k0 + 0 .. k0 + 7]
// elements 8..15 = row[k0 + 16 .. k0 + 23]   (k0 = 0 for lanes 0-15, 8 for lanes 16-31)
static __device__ inline v16bf load_frag_lds(const bf16* rowbase, int k0) {
  v8bf lo = *(const v8bf*)(rowbase + k0);
  v8bf hi = *(const v8bf*)(rowbase + k0 + 16);
  v16bf r;
#pragma unroll
  for (int i = 0; i < 8; ++i) { r[i] = lo[i]; r[i + 8] = hi[i]; }
  return r;
}

// ---------------------------------------------------------------- convert
__global__ void f32_to_bf16_kernel(const float* __restrict__ in,
                                   bf16* __restrict__ out, int n) {
  int stride = gridDim.x * blockDim.x;
  for (int i = blockIdx.x * blockDim.x + threadIdx.x; i < n; i += stride)
    out[i] = (bf16)in[i];
}

// ---------------------------------------------------------------- RoPE
// t is [B, S, nh, HD] bf16, in place; cos/sin are [S, HD/2] fp32
__global__ void rope_kernel(bf16* __restrict__ t,
                            const float* __restrict__ cosp,
                            const float* __restrict__ sinp, int nh) {
  int total = B_ * S_ * nh * (HD_ / 2);
  int stride = gridDim.x * blockDim.x;
  for (int i = blockIdx.x * blockDim.x + threadIdx.x; i < total; i += stride) {
    int p = i & (HD_ / 2 - 1);
    int rest = i >> 6;                 // HD/2 == 64
    int hh = rest % nh; rest /= nh;
    int s = rest % S_;
    int b = rest / S_;
    size_t base = (((size_t)(b * S_ + s) * nh + hh) * HD_) + 2 * p;
    float a  = (float)t[base];
    float bb = (float)t[base + 1];
    float c  = cosp[s * (HD_ / 2) + p];
    float sn = sinp[s * (HD_ / 2) + p];
    t[base]     = (bf16)(a * c - bb * sn);
    t[base + 1] = (bf16)(a * sn + bb * c);
  }
}

// ---------------------------------------------------------------- GEMM
// C[M,N] = A[M,K] @ W[N,K]^T   (A, W bf16 row-major; C bf16 or f32)
// Block tile 64x128, 8 waves (2x4), wave tile 32x32 (2x2 WMMA subtiles).
// K-slabs of 64 staged by the Tensor Data Mover, double-buffered:
// wave 0 issues tensor_load_to_lds for slab t+1 while all waves WMMA slab t.
template <bool OUTF32>
__global__ __launch_bounds__(256) void gemm_bf16_kernel(
    const bf16* __restrict__ A, const bf16* __restrict__ W,
    void* __restrict__ Cout, int M, int N, int K) {
  __shared__ bf16 As[2][64][64];    // 16 KB
  __shared__ bf16 Bs[2][128][64];   // 32 KB

  const int tid   = threadIdx.x;
  const int mBase = blockIdx.y * 64;
  const int nBase = blockIdx.x * 128;
  const int w     = tid >> 5;
  const int lane  = tid & 31;
  const int m16   = lane & 15;
  const int k0    = (lane < 16) ? 0 : 8;
  const int wr    = (w >> 2) * 32;   // 0 / 32
  const int wc    = (w & 3) * 32;    // 0 / 32 / 64 / 96
  const bool wave0 = (tid < 32);

  v8f acc[2][2] = {};

  // Prologue: DMA the first K-slab into buffer 0.
  if (wave0) {
    tdm_load_2d(A + (size_t)mBase * K, &As[0][0][0], 64, 64, (u32)K);
    tdm_load_2d(W + (size_t)nBase * K, &Bs[0][0][0], 128, 64, (u32)K);
    __builtin_amdgcn_s_wait_tensorcnt(0);
  }
  __syncthreads();

  int buf = 0;
  for (int kt = 0; kt < K; kt += 64) {
    // Prefetch next slab into the other buffer via TDM (overlaps with WMMAs).
    if (kt + 64 < K && wave0) {
      tdm_load_2d(A + (size_t)mBase * K + kt + 64, &As[buf ^ 1][0][0], 64, 64, (u32)K);
      tdm_load_2d(W + (size_t)nBase * K + kt + 64, &Bs[buf ^ 1][0][0], 128, 64, (u32)K);
    }

    // Compute on current buffer: 2 K-steps of 32, 4 WMMAs each.
#pragma unroll
    for (int ks = 0; ks < 64; ks += 32) {
      v16bf a0 = load_frag_lds(&As[buf][wr + m16][ks], k0);
      v16bf a1 = load_frag_lds(&As[buf][wr + 16 + m16][ks], k0);
      v16bf b0 = load_frag_lds(&Bs[buf][wc + m16][ks], k0);
      v16bf b1 = load_frag_lds(&Bs[buf][wc + 16 + m16][ks], k0);
      acc[0][0] = __builtin_amdgcn_wmma_f32_16x16x32_bf16(false, a0, false, b0, (short)0, acc[0][0], false, false);
      acc[0][1] = __builtin_amdgcn_wmma_f32_16x16x32_bf16(false, a0, false, b1, (short)0, acc[0][1], false, false);
      acc[1][0] = __builtin_amdgcn_wmma_f32_16x16x32_bf16(false, a1, false, b0, (short)0, acc[1][0], false, false);
      acc[1][1] = __builtin_amdgcn_wmma_f32_16x16x32_bf16(false, a1, false, b1, (short)0, acc[1][1], false, false);
    }

    // Make sure the prefetch landed before anyone reads the other buffer.
    if (wave0) __builtin_amdgcn_s_wait_tensorcnt(0);
    __syncthreads();
    buf ^= 1;
  }

  const int rowHalf = (lane < 16) ? 0 : 8;
#pragma unroll
  for (int ti = 0; ti < 2; ++ti)
#pragma unroll
    for (int tj = 0; tj < 2; ++tj)
#pragma unroll
      for (int i2 = 0; i2 < 8; ++i2) {
        int row = mBase + wr + ti * 16 + i2 + rowHalf;
        int col = nBase + wc + tj * 16 + m16;
        float val = acc[ti][tj][i2];
        if (OUTF32) ((float*)Cout)[(size_t)row * N + col] = val;
        else        ((bf16*)Cout)[(size_t)row * N + col] = (bf16)val;
      }
}

// ---------------------------------------------------------------- flash attention
// One block = (b, h, 128-query slab); 8 waves, each owns a 16-row Q tile.
__global__ __launch_bounds__(256) void flash_attn_kernel(
    const bf16* __restrict__ Q,   // [B, S, H*HD]
    const bf16* __restrict__ Kc,  // [B, S, KVH*HD]
    const bf16* __restrict__ Vc,  // [B, S, KVH*HD]
    bf16* __restrict__ O) {       // [B, S, H*HD]
  __shared__ bf16 Ks[32][128];      // 8 KB  : K chunk, row-major [key][dim]
  __shared__ bf16 VTs[128][32];     // 8 KB  : V chunk transposed [dim][key]
  __shared__ bf16 Qs[8][16][128];   // 32 KB : per-wave Q tile
  __shared__ bf16 Ps[8][16][32];    // 8 KB  : per-wave P re-layout scratch

  const int tiles = S_ / 128;
  const int qt  = blockIdx.x % tiles;
  const int h   = (blockIdx.x / tiles) % H_;
  const int b   = blockIdx.x / (tiles * H_);
  const int kvh = h / NREP_;

  const int tid     = threadIdx.x;
  const int w       = tid >> 5;
  const int lane    = tid & 31;
  const int m16     = lane & 15;
  const int k0      = (lane < 16) ? 0 : 8;
  const int rowHalf = (lane < 16) ? 0 : 8;
  const int qBase   = qt * 128 + w * 16;

  // Load this wave's 16x128 Q tile into LDS (8 x b128 per lane)
  {
    int r = lane >> 1;
    int p = (lane & 1) * 64;
    const bf16* src = Q + ((size_t)(b * S_ + qBase + r)) * (H_ * HD_) + h * HD_ + p;
#pragma unroll
    for (int i = 0; i < 8; ++i)
      *(uint4*)&Qs[w][r][p + i * 8] = *(const uint4*)(src + i * 8);
  }

  v8f acc[8] = {};
  float rmax[8], rsum[8];
#pragma unroll
  for (int i = 0; i < 8; ++i) { rmax[i] = -1e30f; rsum[i] = 0.f; }

  const float scale = 0.08838834764831845f;  // HD^-0.5
  const int kmax = qt * 128 + 128;

  for (int kk = 0; kk < kmax; kk += 32) {
    __syncthreads();
    // Cooperative stage: K chunk row-major, V chunk transposed
    {
      int key = tid >> 3;
      int d16 = (tid & 7) * 16;
      const bf16* ksrc = Kc + ((size_t)(b * S_ + kk + key)) * (KVH_ * HD_) + kvh * HD_ + d16;
      *(uint4*)&Ks[key][d16]     = *(const uint4*)(ksrc);
      *(uint4*)&Ks[key][d16 + 8] = *(const uint4*)(ksrc + 8);
      const bf16* vsrc = Vc + ((size_t)(b * S_ + kk + key)) * (KVH_ * HD_) + kvh * HD_ + d16;
      union { uint4 u[2]; bf16 hh[16]; } tmp;
      tmp.u[0] = *(const uint4*)(vsrc);
      tmp.u[1] = *(const uint4*)(vsrc + 8);
#pragma unroll
      for (int j = 0; j < 16; ++j) VTs[d16 + j][key] = tmp.hh[j];
    }
    __syncthreads();

    // kk and qBase are both multiples of 16, so computed chunks have kk <= qBase
    if (kk <= qBase + 15) {
      // S = Q Kt : two 16x16 score subtiles, K-dim = HD = 4 WMMA steps each
      v8f sc[2] = {};
#pragma unroll
      for (int st = 0; st < 2; ++st) {
#pragma unroll
        for (int kc = 0; kc < HD_; kc += 32) {
          v16bf a  = load_frag_lds(&Qs[w][m16][kc], k0);
          v16bf bb = load_frag_lds(&Ks[st * 16 + m16][kc], k0);
          sc[st] = __builtin_amdgcn_wmma_f32_16x16x32_bf16(false, a, false, bb, (short)0, sc[st], false, false);
        }
      }
      // scale + causal mask (lane holds key column st*16+m16; slot i -> query row)
      float cmax[8], csum[8];
#pragma unroll
      for (int i = 0; i < 8; ++i) {
        int qrow = qBase + i + rowHalf;
#pragma unroll
        for (int st = 0; st < 2; ++st) {
          int key = kk + st * 16 + m16;
          float v = sc[st][i] * scale;
          sc[st][i] = (key <= qrow) ? v : -1e30f;
        }
        cmax[i] = fmaxf(sc[0][i], sc[1][i]);
      }
      // row max across the 16 lanes of each half (xor 1,2,4,8 stays within half)
#pragma unroll
      for (int off = 1; off < 16; off <<= 1)
#pragma unroll
        for (int i = 0; i < 8; ++i)
          cmax[i] = fmaxf(cmax[i], __shfl_xor(cmax[i], off, 32));
      // online softmax update
#pragma unroll
      for (int i = 0; i < 8; ++i) {
        float newm = fmaxf(rmax[i], cmax[i]);
        float f  = __expf(rmax[i] - newm);
        float p0 = __expf(sc[0][i] - newm);
        float p1 = __expf(sc[1][i] - newm);
        sc[0][i] = p0; sc[1][i] = p1;
        csum[i] = p0 + p1;
        rmax[i] = newm;
        rsum[i] *= f;
#pragma unroll
        for (int j = 0; j < 8; ++j) acc[j][i] *= f;
      }
#pragma unroll
      for (int off = 1; off < 16; off <<= 1)
#pragma unroll
        for (int i = 0; i < 8; ++i)
          csum[i] += __shfl_xor(csum[i], off, 32);
#pragma unroll
      for (int i = 0; i < 8; ++i) rsum[i] += csum[i];

      // Re-layout P (C layout -> A fragment layout) through per-wave LDS
#pragma unroll
      for (int i = 0; i < 8; ++i) {
        int r = i + rowHalf;
        Ps[w][r][m16]      = (bf16)sc[0][i];
        Ps[w][r][16 + m16] = (bf16)sc[1][i];
      }
      asm volatile("s_wait_dscnt 0" ::: "memory");  // wave-local LDS RAW fence

      // O += P @ V : K-dim = 32 keys, 8 output column tiles
      v16bf pa = load_frag_lds(&Ps[w][m16][0], k0);
#pragma unroll
      for (int j = 0; j < 8; ++j) {
        v16bf vb = load_frag_lds(&VTs[j * 16 + m16][0], k0);
        acc[j] = __builtin_amdgcn_wmma_f32_16x16x32_bf16(false, pa, false, vb, (short)0, acc[j], false, false);
      }
    }
  }

  // epilogue: normalize and store bf16
#pragma unroll
  for (int i = 0; i < 8; ++i) {
    float inv = 1.0f / rsum[i];
    int row = qBase + i + rowHalf;
#pragma unroll
    for (int j = 0; j < 8; ++j) {
      int col = h * HD_ + j * 16 + m16;
      O[((size_t)(b * S_ + row)) * (H_ * HD_) + col] = (bf16)(acc[j][i] * inv);
    }
  }
}

// ---------------------------------------------------------------- launch
extern "C" void kernel_launch(void* const* d_in, const int* in_sizes, int n_in,
                              void* d_out, int out_size, void* d_ws, size_t ws_size,
                              hipStream_t stream) {
  (void)in_sizes; (void)n_in; (void)out_size; (void)ws_size;
  const float* x    = (const float*)d_in[0];
  const float* wq   = (const float*)d_in[1];
  const float* wk   = (const float*)d_in[2];
  const float* wv   = (const float*)d_in[3];
  const float* wo   = (const float*)d_in[4];
  const float* cosp = (const float*)d_in[5];
  const float* sinp = (const float*)d_in[6];

  char* ws = (char*)d_ws;
  size_t off = 0;
  auto alloc = [&](size_t nbytes) { char* p = ws + off; off += (nbytes + 255) & ~(size_t)255; return p; };
  bf16* xb  = (bf16*)alloc((size_t)B_ * S_ * D_ * 2);
  bf16* wqb = (bf16*)alloc((size_t)H_ * HD_ * D_ * 2);
  bf16* wkb = (bf16*)alloc((size_t)KVH_ * HD_ * D_ * 2);
  bf16* wvb = (bf16*)alloc((size_t)KVH_ * HD_ * D_ * 2);
  bf16* wob = (bf16*)alloc((size_t)D_ * H_ * HD_ * 2);
  bf16* qb  = (bf16*)alloc((size_t)B_ * S_ * H_ * HD_ * 2);
  bf16* kb  = (bf16*)alloc((size_t)B_ * S_ * KVH_ * HD_ * 2);
  bf16* vb  = (bf16*)alloc((size_t)B_ * S_ * KVH_ * HD_ * 2);
  bf16* ab  = (bf16*)alloc((size_t)B_ * S_ * H_ * HD_ * 2);

  auto cvt = [&](const float* in, bf16* out, int n) {
    int blocks = (n + 256 * 8 - 1) / (256 * 8);
    f32_to_bf16_kernel<<<blocks, 256, 0, stream>>>(in, out, n);
  };
  cvt(x,  xb,  B_ * S_ * D_);
  cvt(wq, wqb, H_ * HD_ * D_);
  cvt(wk, wkb, KVH_ * HD_ * D_);
  cvt(wv, wvb, KVH_ * HD_ * D_);
  cvt(wo, wob, D_ * H_ * HD_);

  dim3 blk(256);
  // Q/K/V projections
  gemm_bf16_kernel<false><<<dim3((H_ * HD_) / 128, (B_ * S_) / 64), blk, 0, stream>>>(
      xb, wqb, qb, B_ * S_, H_ * HD_, D_);
  gemm_bf16_kernel<false><<<dim3((KVH_ * HD_) / 128, (B_ * S_) / 64), blk, 0, stream>>>(
      xb, wkb, kb, B_ * S_, KVH_ * HD_, D_);
  gemm_bf16_kernel<false><<<dim3((KVH_ * HD_) / 128, (B_ * S_) / 64), blk, 0, stream>>>(
      xb, wvb, vb, B_ * S_, KVH_ * HD_, D_);

  // RoPE on Q and K (in place)
  rope_kernel<<<(B_ * S_ * H_ * (HD_ / 2) + 255) / 256, 256, 0, stream>>>(qb, cosp, sinp, H_);
  rope_kernel<<<(B_ * S_ * KVH_ * (HD_ / 2) + 255) / 256, 256, 0, stream>>>(kb, cosp, sinp, KVH_);

  // Causal GQA flash attention
  flash_attn_kernel<<<B_ * H_ * (S_ / 128), blk, 0, stream>>>(qb, kb, vb, ab);

  // Output projection, fp32 result
  gemm_bf16_kernel<true><<<dim3(D_ / 128, (B_ * S_) / 64), blk, 0, stream>>>(
      ab, wob, (float*)d_out, B_ * S_, D_, H_ * HD_);
}
